// MultipleContext_66030827208918
// MI455X (gfx1250) — matrix operand order repaired and checked by reading, not compiled, not run on hardware
//
#include <hip/hip_runtime.h>

// ---------------------------------------------------------------------------
// VQ nearest-codebook kernel for MI455X (gfx1250, wave32, WMMA + TDM)
//   argmin_k ( |c_k|^2 - 2 f.c_k )   (|f|^2 is row-constant -> dropped)
//   matmul via v_wmma_f32_16x16x32_f16; codebook staged to LDS by the
//   Tensor Data Mover (tensor_load_to_lds) in its final f16 padded layout.
// ---------------------------------------------------------------------------

typedef _Float16 v16h __attribute__((ext_vector_type(16)));
typedef _Float16 h8   __attribute__((ext_vector_type(8)));
typedef float    v8f  __attribute__((ext_vector_type(8)));
typedef unsigned int v4u  __attribute__((ext_vector_type(4)));
typedef int          v8ii __attribute__((ext_vector_type(8)));
typedef int          v4ii __attribute__((ext_vector_type(4)));

#define Bn 262144
#define Kn 1024
#define Dn 64
#define ROW_H 72                        // padded halves per LDS codebook row
#define BLOB_BYTES (Kn * ROW_H * 2 + Kn * 4)   // f16 table + fp32 norms = 151552
#define BLOB_DW    (BLOB_BYTES / 4)            // 37888 dwords (fits tile_dim0 u16)
#define SMEM_BYTES BLOB_BYTES

// ---- prep: build the LDS image (padded f16 codebook + |c|^2) in d_ws ------
__global__ __launch_bounds__(256) void prep_codebook_kernel(
    const float* __restrict__ emb, void* __restrict__ ws)
{
  const int r = blockIdx.x * 256 + threadIdx.x;   // 4 blocks -> 1024 rows
  if (r >= Kn) return;
  _Float16* wh  = (_Float16*)ws;
  float*    wc2 = (float*)((char*)ws + Kn * ROW_H * 2);
  const float4* src = (const float4*)(emb + r * Dn);
  h8* drow = (h8*)(wh + r * ROW_H);
  float s = 0.f;
#pragma unroll
  for (int i = 0; i < 8; ++i) {
    float4 v0 = src[2 * i + 0];
    float4 v1 = src[2 * i + 1];
    s += v0.x * v0.x + v0.y * v0.y + v0.z * v0.z + v0.w * v0.w
       + v1.x * v1.x + v1.y * v1.y + v1.z * v1.z + v1.w * v1.w;
    h8 h = { (_Float16)v0.x, (_Float16)v0.y, (_Float16)v0.z, (_Float16)v0.w,
             (_Float16)v1.x, (_Float16)v1.y, (_Float16)v1.z, (_Float16)v1.w };
    drow[i] = h;
  }
  wc2[r] = s;
}

// ---------------------------------------------------------------------------
__global__ __launch_bounds__(256) void vq_argmin_kernel(
    const float* __restrict__ inputs,
    const float* __restrict__ emb,
    float* __restrict__ out_nearest,
    int*   __restrict__ out_idx,
    const void* __restrict__ ws_blob,
    int use_tdm)
{
  extern __shared__ char smem[];
  _Float16* embh = (_Float16*)smem;                 // [Kn][ROW_H] f16 codebook
  float*    c2s  = (float*)(smem + Kn * ROW_H * 2); // [Kn] fp32 |c_k|^2

  if (use_tdm) {
    // ---- single TDM DMA of the prebuilt 151552B blob into LDS ----
    if (threadIdx.x < 32) {
      unsigned           lds_base = (unsigned)(size_t)(void*)smem; // LDS byte offset
      unsigned long long ga       = (unsigned long long)(size_t)ws_blob;
      // D# group0: count=1, lds_addr, global_addr[56:0], type=2
      v4u g0 = { 1u,
                 lds_base,
                 (unsigned)(ga & 0xFFFFFFFFu),
                 (unsigned)((ga >> 32) & 0x01FFFFFFu) | 0x80000000u };
      // D# group1: data_size=4B; 1D tile: tensor_dim0=tile_dim0=stride=BLOB_DW
      v8ii g1 = { (int)0x00020000,                    // data_size=2 (4B), mask=0
                  (int)((BLOB_DW & 0xFFFF) << 16),    // tensor_dim0 lo16
                  (int)0x00010000,                    // tensor_dim0 hi=0 | tensor_dim1=1
                  (int)((BLOB_DW & 0xFFFF) << 16),    // tensor_dim1 hi=0 | tile_dim0
                  0,                                  // tile_dim1=0, tile_dim2=0 (1D)
                  (int)BLOB_DW,                       // tensor_dim0_stride lo32
                  0, 0 };
      v4ii z4 = { 0, 0, 0, 0 };
      v8ii z8 = { 0, 0, 0, 0, 0, 0, 0, 0 };
      __builtin_amdgcn_tensor_load_to_lds(g0, g1, z4, z4, z8, 0);
      __builtin_amdgcn_s_wait_tensorcnt(0);
    }
    __syncthreads();
  } else {
    // ---- fallback: stage + convert codebook with plain loads ----
    for (int r = threadIdx.x; r < Kn; r += 256) {
      const float4* src = (const float4*)(emb + r * Dn);
      h8* drow = (h8*)(embh + r * ROW_H);
      float s = 0.f;
#pragma unroll
      for (int i = 0; i < 8; ++i) {
        float4 v0 = src[2 * i + 0];
        float4 v1 = src[2 * i + 1];
        s += v0.x * v0.x + v0.y * v0.y + v0.z * v0.z + v0.w * v0.w
           + v1.x * v1.x + v1.y * v1.y + v1.z * v1.z + v1.w * v1.w;
        h8 h = { (_Float16)v0.x, (_Float16)v0.y, (_Float16)v0.z, (_Float16)v0.w,
                 (_Float16)v1.x, (_Float16)v1.y, (_Float16)v1.z, (_Float16)v1.w };
        drow[i] = h;
      }
      c2s[r] = s;
    }
    __syncthreads();
  }

  const int lane  = threadIdx.x & 31;
  const int wid   = threadIdx.x >> 5;
  const int nlane = lane & 15;
  const int half  = lane >> 4;
  const long rowBase = ((long)blockIdx.x * 8 + wid) * 64;  // 8 waves x 64 rows

  // ---- load A: 4 M-tiles of 16 rows x 64 K in 16-bit 16x32 A layout ----
  v16h A[4][2];
  {
    const int hk = half * 16;
#pragma unroll
    for (int mt = 0; mt < 4; ++mt) {
      const float* fr = inputs + (rowBase + mt * 16 + nlane) * Dn;
#pragma unroll
      for (int c = 0; c < 2; ++c) {
        v16h a;
#pragma unroll
        for (int j = 0; j < 8; ++j) a[j]     = (_Float16)fr[c * 32 + hk + j];
#pragma unroll
        for (int j = 0; j < 8; ++j) a[8 + j] = (_Float16)fr[c * 32 + 8 + hk + j];
        A[mt][c] = a;
      }
    }
  }

  float mn[4][8];
  int   mi[4][8];
#pragma unroll
  for (int mt = 0; mt < 4; ++mt)
#pragma unroll
    for (int v = 0; v < 8; ++v) { mn[mt][v] = 3.4e38f; mi[mt][v] = 0; }

  // ---- sweep 64 N-tiles of 16 codes each ----
  const int kb = half * 16;   // B layout: lanes 0-15 K 0..15, lanes 16-31 K 16..31
  for (int nt = 0; nt < 64; ++nt) {
    const int n = nt * 16 + nlane;
    const _Float16* brow = embh + n * ROW_H;
    h8 b0lo = *(const h8*)(brow + kb);
    h8 b0hi = *(const h8*)(brow + kb + 8);
    h8 b1lo = *(const h8*)(brow + 32 + kb);
    h8 b1hi = *(const h8*)(brow + 32 + kb + 8);
    v16h b0 = __builtin_shufflevector(b0lo, b0hi, 0,1,2,3,4,5,6,7,8,9,10,11,12,13,14,15);
    v16h b1 = __builtin_shufflevector(b1lo, b1hi, 0,1,2,3,4,5,6,7,8,9,10,11,12,13,14,15);
    const float c2n = c2s[n];

#pragma unroll
    for (int mt = 0; mt < 4; ++mt) {
      v8f acc = {};
      acc = __builtin_amdgcn_wmma_f32_16x16x32_f16(false, A[mt][0], false, b0,
                                                   (short)0, acc, false, false);
      acc = __builtin_amdgcn_wmma_f32_16x16x32_f16(false, A[mt][1], false, b1,
                                                   (short)0, acc, false, false);
#pragma unroll
      for (int v = 0; v < 8; ++v) {
        float s = __builtin_fmaf(-2.f, acc[v], c2n);   // |c|^2 - 2 f.c
        if (s < mn[mt][v]) { mn[mt][v] = s; mi[mt][v] = n; }
      }
    }
  }

  // ---- per-row argmin across the 16 lanes holding it, then outputs ----
#pragma unroll
  for (int mt = 0; mt < 4; ++mt) {
#pragma unroll
    for (int v = 0; v < 8; ++v) {
      float m = mn[mt][v];
      int   i = mi[mt][v];
#pragma unroll
      for (int off = 1; off < 16; off <<= 1) {
        float om = __shfl_xor(m, off, 32);
        int   oi = __shfl_xor(i, off, 32);
        if (om < m || (om == m && oi < i)) { m = om; i = oi; }
      }
      const long grow = rowBase + mt * 16 + v + half * 8;
      if (nlane == 0) out_idx[grow] = i;
      const float4* srcr = (const float4*)(emb + (long)i * Dn);
      float4*       dstr = (float4*)(out_nearest + grow * Dn);
      dstr[nlane] = srcr[nlane];
    }
  }
}

// ---------------------------------------------------------------------------
extern "C" void kernel_launch(void* const* d_in, const int* in_sizes, int n_in,
                              void* d_out, int out_size, void* d_ws, size_t ws_size,
                              hipStream_t stream) {
  const float* inputs = (const float*)d_in[0];
  const float* emb    = (const float*)d_in[1];
  float* out = (float*)d_out;

  const size_t BD = (size_t)Bn * Dn;   // 16777216
  const size_t KD = (size_t)Kn * Dn;   // 65536

  // passthrough outputs: inputs copy, emb copy
  (void)hipMemcpyAsync(out,      inputs, BD * sizeof(float), hipMemcpyDeviceToDevice, stream);
  (void)hipMemcpyAsync(out + BD, emb,    KD * sizeof(float), hipMemcpyDeviceToDevice, stream);

  float* out_nearest = out + BD + KD;
  int*   out_idx     = (int*)(out + 2 * BD + KD);

  const int use_tdm = (d_ws != nullptr && ws_size >= (size_t)BLOB_BYTES) ? 1 : 0;
  if (use_tdm) {
    prep_codebook_kernel<<<dim3(4), dim3(256), 0, stream>>>(emb, d_ws);
  }

  (void)hipFuncSetAttribute((const void*)vq_argmin_kernel,
                            hipFuncAttributeMaxDynamicSharedMemorySize, SMEM_BYTES);

  dim3 grid(Bn / 512);   // 8 waves x 64 rows = 512 rows per block
  dim3 block(256);
  vq_argmin_kernel<<<grid, block, SMEM_BYTES, stream>>>(
      inputs, emb, out_nearest, out_idx, d_ws, use_tdm);
}
